// SelfAttention_59674275610839
// MI455X (gfx1250) — compile-verified, matrix operand order checked by reading, not run on hardware
//
#include <hip/hip_runtime.h>

#define NH    16
#define HD    64
#define EMB   1024
#define BATCH 2
#define SEQ   2048
#define QKVE  ((size_t)BATCH * NH * SEQ * HD)   // 4,194,304 elems per tensor

typedef __attribute__((ext_vector_type(16))) __bf16 v16bf;
typedef __attribute__((ext_vector_type(8)))  __bf16 v8bf;
typedef __attribute__((ext_vector_type(8)))  float  v8f;

__device__ inline v8f wmma_bf16(v16bf a, v16bf b, v8f c) {
  // D = A(16x32 bf16) * B(32x16 bf16) + C(16x16 f32)
  return __builtin_amdgcn_wmma_f32_16x16x32_bf16(false, a, false, b, (short)0, c,
                                                 false, false);
}

__device__ inline v8f vzero8() {
  v8f z = {0.f, 0.f, 0.f, 0.f, 0.f, 0.f, 0.f, 0.f};
  return z;
}

// Load a 16-element bf16 A/B fragment from a row with K contiguous in memory.
// Position i holds K = base + (i/8)*16 + half*8 + (i%8)  (wave32 16-bit layout).
__device__ inline v16bf ld_bf16_frag(const __bf16* row, int base, int half) {
  v16bf d;
#pragma unroll
  for (int g = 0; g < 2; ++g) {
    v8bf h = *(const v8bf*)(row + base + g * 16 + half * 8);
#pragma unroll
    for (int j = 0; j < 8; ++j) d[g * 8 + j] = h[j];
  }
  return d;
}

// Same fragment layout, but source row is f32 (converted to bf16 on the fly).
__device__ inline v16bf ld_f32_frag(const float* row, int base, int half) {
  v16bf d;
#pragma unroll
  for (int g = 0; g < 2; ++g) {
    const float* p = row + base + g * 16 + half * 8;
    float4 x = *(const float4*)p;
    float4 y = *(const float4*)(p + 4);
    d[g * 8 + 0] = (__bf16)x.x; d[g * 8 + 1] = (__bf16)x.y;
    d[g * 8 + 2] = (__bf16)x.z; d[g * 8 + 3] = (__bf16)x.w;
    d[g * 8 + 4] = (__bf16)y.x; d[g * 8 + 5] = (__bf16)y.y;
    d[g * 8 + 6] = (__bf16)y.z; d[g * 8 + 7] = (__bf16)y.w;
  }
  return d;
}

// ---------------------------------------------------------------------------
// Kernel A: per-head projections. out = X(16xHD) @ W^T(HDxHD), bf16 result.
// ---------------------------------------------------------------------------
__device__ inline void project_head(const float* __restrict__ xrow,
                                    const float* __restrict__ W,
                                    __bf16* __restrict__ qk_out,
                                    __bf16* __restrict__ t_out,
                                    int s0, int lane) {
  const int half = lane >> 4, n = lane & 15;
  v16bf a0 = ld_f32_frag(xrow, 0, half);   // A: rows = seq, K = d (0..31)
  v16bf a1 = ld_f32_frag(xrow, 32, half);  // K = d (32..63)
#pragma unroll
  for (int ct = 0; ct < 4; ++ct) {
    const float* wrow = W + (size_t)(ct * 16 + n) * HD;  // B[d][c] = W[c][d]
    v16bf b0 = ld_f32_frag(wrow, 0, half);
    v16bf b1 = ld_f32_frag(wrow, 32, half);
    v8f acc = vzero8();
    acc = wmma_bf16(a0, b0, acc);
    acc = wmma_bf16(a1, b1, acc);
    if (qk_out) {  // D: lane col c=ct*16+n, rows s = r + 8*half
#pragma unroll
      for (int r = 0; r < 8; ++r)
        qk_out[(size_t)(r + 8 * half) * HD + ct * 16 + n] = (__bf16)acc[r];
    } else {       // transposed store: [d][s], contiguous along s
      v8bf pk;
#pragma unroll
      for (int r = 0; r < 8; ++r) pk[r] = (__bf16)acc[r];
      *(v8bf*)(t_out + (size_t)(ct * 16 + n) * SEQ + s0 + 8 * half) = pk;
    }
  }
}

__global__ void __launch_bounds__(256) qkv_proj_kernel(
    const float* __restrict__ vals, const float* __restrict__ keys,
    const float* __restrict__ qrys, const float* __restrict__ Wq,
    const float* __restrict__ Wk, const float* __restrict__ Wv,
    __bf16* __restrict__ Qb, __bf16* __restrict__ Kb, __bf16* __restrict__ Vt) {
  const int wid  = blockIdx.x * 8 + (threadIdx.x >> 5);
  const int lane = threadIdx.x & 31;
  const int head = wid & (NH - 1);
  const int rt   = wid >> 4;              // row-tile over B*SEQ/16
  const int b    = rt >> 7;               // / (SEQ/16)
  const int s0   = (rt & 127) << 4;
  const int n    = lane & 15;
  const size_t bh   = (size_t)b * NH + head;
  const size_t xoff = ((size_t)b * SEQ + s0 + n) * EMB + head * HD;

  project_head(qrys + xoff, Wq, Qb + (bh * SEQ + s0) * HD, nullptr, s0, lane);
  project_head(keys + xoff, Wk, Kb + (bh * SEQ + s0) * HD, nullptr, s0, lane);
  project_head(vals + xoff, Wv, nullptr, Vt + bh * (size_t)HD * SEQ, s0, lane);
}

// ---------------------------------------------------------------------------
// Kernel B: flash attention. Block = 8 waves sharing one (batch, head);
// K/V^T tiles (32 keys) staged in LDS once per block, 8x L2-traffic cut.
// Per wave: S^T = K*Q^T (D-frag == B-frag layout of P^T), O^T += V^T*P^T.
// ---------------------------------------------------------------------------
__global__ void __launch_bounds__(256) attn_kernel(
    const __bf16* __restrict__ Qb, const __bf16* __restrict__ Kb,
    const __bf16* __restrict__ Vt, const int* __restrict__ mask,
    __bf16* __restrict__ AO) {
  __shared__ __bf16 sK[32 * HD];   // 32 keys x 64 d   (4 KB)
  __shared__ __bf16 sV[HD * 32];   // 64 d   x 32 keys (4 KB)

  const int tid  = threadIdx.x;
  const int lane = tid & 31;
  const int wave = tid >> 5;
  const int half = lane >> 4, n = lane & 15;

  // grid: 512 blocks = B(2) x NH(16) x 16 query-groups of 128
  const int qg   = blockIdx.x & 15;
  const int head = (blockIdx.x >> 4) & (NH - 1);
  const int b    = blockIdx.x >> 8;
  const int q0   = qg * 128 + wave * 16;
  const size_t bh = (size_t)b * NH + head;

  const __bf16* kbase = Kb + bh * (size_t)SEQ * HD;
  const __bf16* vbase = Vt + bh * (size_t)HD * SEQ;
  const int*    mbase = mask + (size_t)b * SEQ;

  // cooperative-load coordinates (16B per thread per tile)
  const int krow = tid >> 3, kc8 = (tid & 7) * 8;   // 32 rows x 64 cols
  const int vrow_c = tid >> 2, vc8 = (tid & 3) * 8; // 64 rows x 32 cols

  // Q as B-operand: lane n = query column, K-dim = d contiguous.
  const __bf16* qrow = Qb + (bh * SEQ + q0 + n) * (size_t)HD;
  v16bf bq0 = ld_bf16_frag(qrow, 0, half);
  v16bf bq1 = ld_bf16_frag(qrow, 32, half);

  v8f o0 = vzero8(), o1 = vzero8(), o2 = vzero8(), o3 = vzero8();
  float m_run = -1e30f, l_run = 0.f;
  const float scale = 0.03125f;  // 1/sqrt(EMB) = 1/32

  for (int k0 = 0; k0 < SEQ; k0 += 32) {
    __syncthreads();  // previous iteration's LDS reads complete
    *(v8bf*)(sK + krow * HD + kc8) =
        *(const v8bf*)(kbase + (size_t)(k0 + krow) * HD + kc8);
    *(v8bf*)(sV + vrow_c * 32 + vc8) =
        *(const v8bf*)(vbase + (size_t)vrow_c * SEQ + k0 + vc8);
    if (k0 + 32 < SEQ) {
      __builtin_prefetch(kbase + (size_t)(k0 + 32 + krow) * HD + kc8, 0, 1);
      __builtin_prefetch(vbase + (size_t)vrow_c * SEQ + k0 + 32 + vc8, 0, 1);
    }
    __syncthreads();  // tiles visible to all waves

    // scores^T from LDS: D[k_local][q]
    v8f st0 = vzero8(), st1 = vzero8();
    st0 = wmma_bf16(ld_bf16_frag(sK + (size_t)n * HD, 0, half), bq0, st0);
    st0 = wmma_bf16(ld_bf16_frag(sK + (size_t)n * HD, 32, half), bq1, st0);
    st1 = wmma_bf16(ld_bf16_frag(sK + (size_t)(16 + n) * HD, 0, half), bq0, st1);
    st1 = wmma_bf16(ld_bf16_frag(sK + (size_t)(16 + n) * HD, 32, half), bq1, st1);

    // mask rows this lane owns: k = k0 + t*16 + 8*half + r
    const int* mrow0 = mbase + k0 + 8 * half;
    int4 ma = *(const int4*)mrow0;
    int4 mb = *(const int4*)(mrow0 + 4);
    int4 mc = *(const int4*)(mrow0 + 16);
    int4 md = *(const int4*)(mrow0 + 20);
    int mk0[8] = {ma.x, ma.y, ma.z, ma.w, mb.x, mb.y, mb.z, mb.w};
    int mk1[8] = {mc.x, mc.y, mc.z, mc.w, md.x, md.y, md.z, md.w};

    float p0[8], p1[8];
    float mx = -3.0e38f;
#pragma unroll
    for (int r = 0; r < 8; ++r) {
      float v0 = mk0[r] ? st0[r] * scale : -1.0e28f;
      float v1 = mk1[r] ? st1[r] * scale : -1.0e28f;
      p0[r] = v0; p1[r] = v1;
      mx = fmaxf(mx, fmaxf(v0, v1));
    }
    mx = fmaxf(mx, __shfl_xor(mx, 16, 32));  // both key-halves of this query
    float mnew  = fmaxf(m_run, mx);
    float alpha = __expf(m_run - mnew);
    float rsum  = 0.f;
    v16bf bp;  // P^T straight into B-fragment layout
#pragma unroll
    for (int r = 0; r < 8; ++r) {
      float e0 = __expf(p0[r] - mnew);
      float e1 = __expf(p1[r] - mnew);
      rsum += e0 + e1;
      bp[r]     = (__bf16)e0;
      bp[r + 8] = (__bf16)e1;
    }
    rsum += __shfl_xor(rsum, 16, 32);
    l_run = l_run * alpha + rsum;
    m_run = mnew;
    o0 *= alpha; o1 *= alpha; o2 *= alpha; o3 *= alpha;

    // O^T += V^T * P^T  (A rows = d, K contiguous along keys in sV)
    o0 = wmma_bf16(ld_bf16_frag(sV + (size_t)(0 * 16 + n) * 32, 0, half), bp, o0);
    o1 = wmma_bf16(ld_bf16_frag(sV + (size_t)(1 * 16 + n) * 32, 0, half), bp, o1);
    o2 = wmma_bf16(ld_bf16_frag(sV + (size_t)(2 * 16 + n) * 32, 0, half), bp, o2);
    o3 = wmma_bf16(ld_bf16_frag(sV + (size_t)(3 * 16 + n) * 32, 0, half), bp, o3);
  }

  const float inv_l = 1.0f / l_run;
  // store O (un-transposed on the fly): lane owns query q0+n, 8 contiguous d
  __bf16* arow = AO + ((size_t)b * SEQ + q0 + n) * EMB + head * HD + 8 * half;
#define STORE_O(oN, dt)                                          \
  {                                                              \
    v8bf pk;                                                     \
    _Pragma("unroll") for (int r = 0; r < 8; ++r)                \
        pk[r] = (__bf16)(oN[r] * inv_l);                         \
    *(v8bf*)(arow + (dt) * 16) = pk;                             \
  }
  STORE_O(o0, 0) STORE_O(o1, 1) STORE_O(o2, 2) STORE_O(o3, 3)
#undef STORE_O
}

// ---------------------------------------------------------------------------
// Kernel C: Y = AO @ Wout^T + bout, f32 output. 16x64 tile per wave.
// ---------------------------------------------------------------------------
__global__ void __launch_bounds__(256) out_proj_kernel(
    const __bf16* __restrict__ AO, const float* __restrict__ Wout,
    const float* __restrict__ bout, float* __restrict__ Y) {
  const int wid  = blockIdx.x * 8 + (threadIdx.x >> 5);
  const int lane = threadIdx.x & 31;
  const int half = lane >> 4, n = lane & 15;
  const int rt = wid & 255;   // row tile over B*SEQ/16
  const int cg = wid >> 8;    // 64-wide column group
  const int s0 = rt << 4;
  const int e0 = cg << 6;

  const __bf16* arow = AO + (size_t)(s0 + n) * EMB;
  v8f acc0 = vzero8(), acc1 = vzero8(), acc2 = vzero8(), acc3 = vzero8();

  for (int kk = 0; kk < EMB; kk += 32) {
    v16bf a = ld_bf16_frag(arow, kk, half);
    acc0 = wmma_bf16(a, ld_f32_frag(Wout + (size_t)(e0 + 0  + n) * EMB, kk, half), acc0);
    acc1 = wmma_bf16(a, ld_f32_frag(Wout + (size_t)(e0 + 16 + n) * EMB, kk, half), acc1);
    acc2 = wmma_bf16(a, ld_f32_frag(Wout + (size_t)(e0 + 32 + n) * EMB, kk, half), acc2);
    acc3 = wmma_bf16(a, ld_f32_frag(Wout + (size_t)(e0 + 48 + n) * EMB, kk, half), acc3);
  }

  const float b0 = bout[e0 + n],      b1 = bout[e0 + 16 + n];
  const float b2 = bout[e0 + 32 + n], b3 = bout[e0 + 48 + n];
#pragma unroll
  for (int r = 0; r < 8; ++r) {
    const size_t row = (size_t)(s0 + r + 8 * half) * EMB + e0 + n;
    Y[row +  0] = acc0[r] + b0;
    Y[row + 16] = acc1[r] + b1;
    Y[row + 32] = acc2[r] + b2;
    Y[row + 48] = acc3[r] + b3;
  }
}

// ---------------------------------------------------------------------------
extern "C" void kernel_launch(void* const* d_in, const int* in_sizes, int n_in,
                              void* d_out, int out_size, void* d_ws, size_t ws_size,
                              hipStream_t stream) {
  const float* vals = (const float*)d_in[0];
  const float* keys = (const float*)d_in[1];
  const float* qrys = (const float*)d_in[2];
  const int*   mask = (const int*)d_in[3];
  const float* Wq   = (const float*)d_in[4];
  const float* Wk   = (const float*)d_in[5];
  const float* Wv   = (const float*)d_in[6];
  const float* Wout = (const float*)d_in[7];
  const float* bout = (const float*)d_in[8];
  float* Y = (float*)d_out;

  __bf16* Qb = (__bf16*)d_ws;
  __bf16* Kb = Qb + QKVE;
  __bf16* Vt = Kb + QKVE;
  __bf16* AO = Vt + QKVE;

  qkv_proj_kernel<<<512, 256, 0, stream>>>(vals, keys, qrys, Wq, Wk, Wv, Qb, Kb, Vt);
  attn_kernel<<<512, 256, 0, stream>>>(Qb, Kb, Vt, mask, AO);
  out_proj_kernel<<<512, 256, 0, stream>>>(AO, Wout, bout, Y);
}
